// GraphEncoder_88484916232488
// MI455X (gfx1250) — compile-verified, matrix operand order checked by reading, not run on hardware
//
#include <hip/hip_runtime.h>

#define Nn 384
#define NODE_DIM 128
#define EDGE_DIM 64
#define HID 256
#define NN (Nn*Nn)
#define THR 0.3f

typedef _Float16 v16h __attribute__((ext_vector_type(16)));
typedef _Float16 v8h  __attribute__((ext_vector_type(8)));
typedef float    v8f  __attribute__((ext_vector_type(8)));
typedef float    v4f  __attribute__((ext_vector_type(4)));

__device__ __forceinline__ v8f wmma_f16(v16h a, v16h b, v8f c) {
  return __builtin_amdgcn_wmma_f32_16x16x32_f16(false, a, false, b, (short)0, c, false, false);
}

// Pack a KxNcols f32 matrix into CDNA5 WMMA B-fragment layout (f16).
// Fragment linear index: (((nt*(K/32) + kt)*32 + lane)*16 + h)
// lane<16: col = nt*16+lane, K = kt*32 + {0..7, 16..23}
// lane>=16: col = nt*16+lane-16, K = kt*32 + {8..15, 24..31}
__global__ void pack_b_kernel(_Float16* __restrict__ dst, const float* __restrict__ src,
                              int K, int Ncols, int total) {
  int idx = blockIdx.x * blockDim.x + threadIdx.x;
  if (idx >= total) return;
  int h  = idx & 15;
  int l  = (idx >> 4) & 31;
  int rest = idx >> 9;
  int KT = K >> 5;
  int kt = rest % KT;
  int nt = rest / KT;
  int col = nt * 16 + (l & 15);
  int hi  = l >> 4;
  int kk  = kt * 32 + hi * 8 + ((h < 8) ? h : h + 8);
  dst[idx] = (_Float16)src[kk * Ncols + col];
}

// Edge encoder: ef[j,i] = relu(ef_raw[j,i]@E1+b1) @ E2 + b2, stored transposed as
// ef_t[i*N+j][64] in f16. Also emits wdT[i*N+j]=wd[j,i], wb[i*N+j], mask_e[i*N+j].
// One wave handles 16 edges (fixed i, contiguous j). Layer-2 GEMM via WMMA.
__global__ __launch_bounds__(256)
void edge_enc_kernel(const float* __restrict__ W, const float* __restrict__ Mm,
                     const float* __restrict__ E1, const float* __restrict__ b1,
                     const _Float16* __restrict__ E2p, const float* __restrict__ b2,
                     _Float16* __restrict__ ef_t, float* __restrict__ wdT,
                     float* __restrict__ wb, float* __restrict__ mask_e) {
  __shared__ float xs[8][16], ss[8][16];
  __shared__ _Float16 tbuf[8][1024] __attribute__((aligned(16)));
  int tid = threadIdx.x;
  int w = tid >> 5, l = tid & 31, lo = l & 15, hi = l >> 4;
  int gw = blockIdx.x * 8 + w;            // global wave -> 16 edges
  int i  = (gw * 16) / Nn;
  int jb = (gw * 16) % Nn;
  if (hi == 0) {
    int jj = jb + lo;
    float Wji = W[jj * Nn + i], Mji = Mm[jj * Nn + i];
    float Wij = W[i * Nn + jj], Mij = Mm[i * Nn + jj];
    float aw = fabsf(Wji), am = fabsf(Mji);
    xs[w][lo] = aw + 0.5f * am;
    ss[w][lo] = (Wji > 0.f) ? 1.f : ((Wji < 0.f) ? -1.f : 0.f);
    wdT[i * Nn + jj] = (aw > THR) ? aw : 0.f;
    float amij = fabsf(Mij);
    wb[i * Nn + jj] = (amij > THR && i != jj) ? amij : 0.f;
    mask_e[i * Nn + jj] = ((fabsf(Wij) > THR) || (amij > THR)) ? 1.f : 0.f;
  }
  __syncthreads();
  // MLP layer 1: t[e][k] = relu(x*E1[0][k] + s*E1[1][k] + b1[k]) -> LDS f16
  for (int it = 0; it < 32; ++it) {
    int idx = it * 32 + l;
    int e = idx >> 6, k = idx & 63;
    float t = xs[w][e] * E1[k] + ss[w][e] * E1[64 + k] + b1[k];
    tbuf[w][idx] = (_Float16)fmaxf(t, 0.f);
  }
  __syncthreads();
  // A fragments (16 edges x K=64, two k-steps)
  v16h afr[2];
  for (int ks = 0; ks < 2; ++ks) {
    int base = lo * 64 + ks * 32 + hi * 8;
    v8h p0 = *(const v8h*)&tbuf[w][base];
    v8h p1 = *(const v8h*)&tbuf[w][base + 16];
    for (int h = 0; h < 8; ++h) { afr[ks][h] = p0[h]; afr[ks][8 + h] = p1[h]; }
  }
  const v16h* Bp = (const v16h*)E2p;
  for (int nt = 0; nt < 4; ++nt) {
    v8f c = {};
    c = wmma_f16(afr[0], Bp[(nt * 2 + 0) * 32 + l], c);
    c = wmma_f16(afr[1], Bp[(nt * 2 + 1) * 32 + l], c);
    int col = nt * 16 + lo;
    float bias = b2[col];
    for (int r = 0; r < 8; ++r) {
      int row = gw * 16 + r + hi * 8;     // == i*N + j
      ef_t[(size_t)row * 64 + col] = (_Float16)(c[r] + bias);
    }
  }
}

// Per-layer precompute: a_s=h@As, at=h@At, hBs=h@Bs, hBt=h@Bt, h_self=MLP2(h,node)
__global__ __launch_bounds__(256)
void pre_kernel(const float* __restrict__ h, int ind,
                const float* __restrict__ As, const float* __restrict__ At,
                const float* __restrict__ Bs, const float* __restrict__ Bt,
                const float* __restrict__ nW1, const float* __restrict__ nb1,
                const float* __restrict__ nW2, const float* __restrict__ nb2,
                float* __restrict__ a_s, float* __restrict__ at_,
                float* __restrict__ hBs, float* __restrict__ hBt,
                float* __restrict__ h_self) {
  __shared__ float hrow[HID];
  __shared__ float hs1[HID];
  int i = blockIdx.x, t = threadIdx.x;
  if (t < ind) hrow[t] = h[i * ind + t];
  __syncthreads();
  float a0 = 0, a1 = 0, a2 = 0, a3 = 0, an = 0;
  for (int k = 0; k < ind; ++k) {
    float hv = hrow[k];
    a0 += hv * As[k * HID + t];
    a1 += hv * At[k * HID + t];
    a2 += hv * Bs[k * HID + t];
    a3 += hv * Bt[k * HID + t];
    an += hv * nW1[k * HID + t];
  }
  a_s[i * HID + t] = a0; at_[i * HID + t] = a1;
  hBs[i * HID + t] = a2; hBt[i * HID + t] = a3;
  hs1[t] = fmaxf(an + nb1[t], 0.f);
  __syncthreads();
  float hs = nb2[t];
  for (int k = 0; k < HID; ++k) hs += hs1[k] * nW2[k * HID + t];
  h_self[i * HID + t] = hs;
}

// One block per node i. WMMA computes ef@Ae tiles fused with relu + wd-weighted
// reduction over j. bmsg branch is pure VALU. Then GEMVs + layernorm + relu.
__global__ __launch_bounds__(256)
void layer_kernel(const _Float16* __restrict__ ef_t, const _Float16* __restrict__ Aep,
                  const float* __restrict__ wdT, const float* __restrict__ wbp,
                  const float* __restrict__ a_s, const float* __restrict__ at_,
                  const float* __restrict__ hBs, const float* __restrict__ hBt,
                  const float* __restrict__ h_self,
                  const float* __restrict__ bd, const float* __restrict__ bb,
                  const float* __restrict__ l2dW, const float* __restrict__ l2db,
                  const float* __restrict__ l2bW, const float* __restrict__ l2bb,
                  const float* __restrict__ lng, const float* __restrict__ lnb,
                  float* __restrict__ h_out, float* __restrict__ out_node) {
  __shared__ float Sw[8][HID];
  __shared__ float SB[HID];
  __shared__ float cdl[HID], cbl[HID];
  __shared__ float red[HID];
  int i = blockIdx.x, t = threadIdx.x;
  int w = t >> 5, l = t & 31, lo = l & 15, hi = l >> 4;
  cdl[t] = at_[i * HID + t] + bd[t];
  cbl[t] = hBt[i * HID + t] + bb[t];
  for (int ww = 0; ww < 8; ++ww) Sw[ww][t] = 0.f;
  __syncthreads();
  // sum of wd / wb over j (for folded bias terms)
  float psd = 0.f, psb = 0.f;
  for (int j = t; j < Nn; j += HID) { psd += wdT[i * Nn + j]; psb += wbp[i * Nn + j]; }
  red[t] = psd; __syncthreads();
  for (int s = 128; s > 0; s >>= 1) { if (t < s) red[t] += red[t + s]; __syncthreads(); }
  float sumwd = red[0]; __syncthreads();
  red[t] = psb; __syncthreads();
  for (int s = 128; s > 0; s >>= 1) { if (t < s) red[t] += red[t + s]; __syncthreads(); }
  float sumwb = red[0]; __syncthreads();
  // bmsg: S_b[d] = sum_j wb[i,j] * relu(hBs[j][d] + hBt[i][d] + bb[d])
  float accb = 0.f;
  float cb = cbl[t];
  for (int j = 0; j < Nn; ++j)
    accb += wbp[i * Nn + j] * fmaxf(hBs[j * HID + t] + cb, 0.f);
  SB[t] = accb;
  // dmsg: WMMA over j-tiles, fused relu + wd-weighted accumulate
  float sacc[16];
  for (int nt = 0; nt < 16; ++nt) sacc[nt] = 0.f;
  const v16h* Bp = (const v16h*)Aep;
  for (int jt = w; jt < 24; jt += 8) {
    v16h afr[2];
    const _Float16* rowp = ef_t + (size_t)(i * Nn + jt * 16 + lo) * 64;
    for (int ks = 0; ks < 2; ++ks) {
      v8h p0 = *(const v8h*)(rowp + ks * 32 + hi * 8);
      v8h p1 = *(const v8h*)(rowp + ks * 32 + hi * 8 + 16);
      for (int h = 0; h < 8; ++h) { afr[ks][h] = p0[h]; afr[ks][8 + h] = p1[h]; }
    }
    float wdv[8];
    for (int r = 0; r < 8; ++r) wdv[r] = wdT[i * Nn + jt * 16 + r + hi * 8];
    for (int nt = 0; nt < 16; ++nt) {
      v8f c = {};
      c = wmma_f16(afr[0], Bp[(nt * 2 + 0) * 32 + l], c);
      c = wmma_f16(afr[1], Bp[(nt * 2 + 1) * 32 + l], c);
      int d = nt * 16 + lo;
      float cdd = cdl[d];
      float s = 0.f;
      for (int r = 0; r < 8; ++r) {
        int j = jt * 16 + r + hi * 8;
        s += wdv[r] * fmaxf(c[r] + a_s[j * HID + d] + cdd, 0.f);
      }
      sacc[nt] += s;
    }
  }
  for (int nt = 0; nt < 16; ++nt) {
    float v = sacc[nt] + __shfl_xor(sacc[nt], 16, 32);  // combine hi halves (wave32)
    if (hi == 0) Sw[w][nt * 16 + lo] = v;
  }
  __syncthreads();
  float Sd = 0.f;
  for (int ww = 0; ww < 8; ++ww) Sd += Sw[ww][t];
  red[t] = Sd;
  __syncthreads();
  // GEMVs: h_dir = S_d@W2d + sumwd*b2d ; h_bid = S_b@W2b + sumwb*b2b
  float hd = sumwd * l2db[t];
  float hb = sumwb * l2bb[t];
  for (int k = 0; k < HID; ++k) {
    hd += red[k] * l2dW[k * HID + t];
    hb += SB[k] * l2bW[k * HID + t];
  }
  float x = h_self[i * HID + t] + hd + hb;
  __syncthreads();
  // layernorm + relu
  red[t] = x; __syncthreads();
  for (int s = 128; s > 0; s >>= 1) { if (t < s) red[t] += red[t + s]; __syncthreads(); }
  float mean = red[0] / HID; __syncthreads();
  float dx = x - mean;
  red[t] = dx * dx; __syncthreads();
  for (int s = 128; s > 0; s >>= 1) { if (t < s) red[t] += red[t + s]; __syncthreads(); }
  float var = red[0] / HID; __syncthreads();
  float y = fmaxf(dx * rsqrtf(var + 1e-5f) * lng[t] + lnb[t], 0.f);
  h_out[i * HID + t] = y;
  if (out_node) out_node[i * HID + t] = y;
}

// edge_emb[i,j] = concat(h[i], h[j]) * mask_e[i,j]  (302 MB, pure bandwidth; NT stores)
__global__ __launch_bounds__(256)
void edge_emb_kernel(const float* __restrict__ h, const float* __restrict__ mask_e,
                     float* __restrict__ out) {
  long idx = (long)blockIdx.x * 256 + threadIdx.x;   // one v4f each
  long e = idx >> 7;                                  // 128 v4f per edge
  int c4 = (int)(idx & 127) << 2;
  int i = (int)(e / Nn), j = (int)(e % Nn);
  float m = mask_e[e];
  const float* src = (c4 < HID) ? (h + i * HID + c4) : (h + j * HID + (c4 - HID));
  v4f v = *(const v4f*)src;
  v *= m;
  __builtin_nontemporal_store(v, (v4f*)out + idx);
}

__global__ __launch_bounds__(256)
void graph_kernel(const float* __restrict__ h, const float* __restrict__ W1,
                  const float* __restrict__ b1, const float* __restrict__ W2,
                  const float* __restrict__ b2, float* __restrict__ out) {
  __shared__ float m[HID], r1[HID];
  int t = threadIdx.x;
  float s = 0.f;
  for (int i = 0; i < Nn; ++i) s += h[i * HID + t];
  m[t] = s / (float)Nn;
  __syncthreads();
  float a = b1[t];
  for (int k = 0; k < HID; ++k) a += m[k] * W1[k * HID + t];
  r1[t] = fmaxf(a, 0.f);
  __syncthreads();
  float o = b2[t];
  for (int k = 0; k < HID; ++k) o += r1[k] * W2[k * HID + t];
  out[t] = o;
}

extern "C" void kernel_launch(void* const* d_in, const int* in_sizes, int n_in,
                              void* d_out, int out_size, void* d_ws, size_t ws_size,
                              hipStream_t stream) {
  (void)in_sizes; (void)n_in; (void)out_size; (void)ws_size;
  const float* node = (const float*)d_in[0];
  const float* W    = (const float*)d_in[1];
  const float* Mm   = (const float*)d_in[2];
  const float* eeW1 = (const float*)d_in[3];
  const float* eeB1 = (const float*)d_in[4];
  const float* eeW2 = (const float*)d_in[5];
  const float* eeB2 = (const float*)d_in[6];
  // per-layer leaves, jax tree order (dict keys sorted): bmsg(Bs,Bt,bb,W2,b2),
  // dmsg(As,At,Ae,bd,W2,b2), ln(g,b), node(W1,b1,W2,b2)
  auto L = [&](int layer, int off) { return (const float*)d_in[7 + layer * 17 + off]; };
  const float* roW1 = (const float*)d_in[58];
  const float* roB1 = (const float*)d_in[59];
  const float* roW2 = (const float*)d_in[60];
  const float* roB2 = (const float*)d_in[61];

  char* ws = (char*)d_ws;
  size_t off = 0;
  auto alloc = [&](size_t bytes) { void* p = ws + off; off = (off + bytes + 255) & ~(size_t)255; return p; };
  _Float16* ef_t  = (_Float16*)alloc((size_t)NN * 64 * 2);
  float* wdT      = (float*)alloc((size_t)NN * 4);
  float* wb       = (float*)alloc((size_t)NN * 4);
  float* mask_e   = (float*)alloc((size_t)NN * 4);
  float* a_s      = (float*)alloc((size_t)Nn * HID * 4);
  float* at_      = (float*)alloc((size_t)Nn * HID * 4);
  float* hBs      = (float*)alloc((size_t)Nn * HID * 4);
  float* hBt      = (float*)alloc((size_t)Nn * HID * 4);
  float* h_self   = (float*)alloc((size_t)Nn * HID * 4);
  float* hbA      = (float*)alloc((size_t)Nn * HID * 4);
  float* hbB      = (float*)alloc((size_t)Nn * HID * 4);
  _Float16* E2p   = (_Float16*)alloc(4096 * 2);
  _Float16* Aep[3];
  for (int l = 0; l < 3; ++l) Aep[l] = (_Float16*)alloc(16384 * 2);

  // pack WMMA B operands
  pack_b_kernel<<<16, 256, 0, stream>>>(E2p, eeW2, 64, 64, 4096);
  for (int l = 0; l < 3; ++l)
    pack_b_kernel<<<64, 256, 0, stream>>>(Aep[l], L(l, 7), 64, 256, 16384);

  // edge features (WMMA) + masks
  edge_enc_kernel<<<NN / 128, 256, 0, stream>>>(W, Mm, eeW1, eeB1, E2p, eeB2,
                                                ef_t, wdT, wb, mask_e);

  const float* hcur = node;
  float* houts[3] = { hbA, hbB, hbA };
  float* out = (float*)d_out;
  for (int l = 0; l < 3; ++l) {
    int ind = (l == 0) ? NODE_DIM : HID;
    pre_kernel<<<Nn, HID, 0, stream>>>(hcur, ind, L(l, 5), L(l, 6), L(l, 0), L(l, 1),
                                       L(l, 13), L(l, 14), L(l, 15), L(l, 16),
                                       a_s, at_, hBs, hBt, h_self);
    layer_kernel<<<Nn, HID, 0, stream>>>(ef_t, Aep[l], wdT, wb, a_s, at_, hBs, hBt, h_self,
                                         L(l, 8), L(l, 2), L(l, 9), L(l, 10), L(l, 3), L(l, 4),
                                         L(l, 11), L(l, 12),
                                         houts[l], (l == 2) ? out : nullptr);
    hcur = houts[l];
  }
  edge_emb_kernel<<<NN * 128 / 256, 256, 0, stream>>>(hbA, mask_e, out + Nn * HID);
  graph_kernel<<<1, 256, 0, stream>>>(hbA, roW1, roB1, roW2, roB2,
                                      out + Nn * HID + (size_t)NN * 2 * HID);
}
